// MoEFFNLayer_6940667150378
// MI455X (gfx1250) — matrix operand order repaired and checked by reading, not compiled, not run on hardware
//
#include <hip/hip_runtime.h>
#include <hip/hip_bf16.h>

typedef __attribute__((ext_vector_type(16))) __bf16 v16bf;
typedef __attribute__((ext_vector_type(8)))  __bf16 v8bf;
typedef __attribute__((ext_vector_type(8)))  float  v8f;
typedef __attribute__((ext_vector_type(4)))  float  v4f;

#define DMODEL 1024
#define NEXP   8
#define HID    4096

// ---------------- router: logits, top-2, softmax, expert counts ----------------
__global__ __launch_bounds__(256) void moe_router(const float* __restrict__ x,
                                                  const float* __restrict__ gate_w,
                                                  int* __restrict__ counts,
                                                  int* __restrict__ tkIdx,
                                                  float* __restrict__ tkW, int T) {
  __shared__ float gw[NEXP * DMODEL];
  for (int i = threadIdx.x; i < NEXP * DMODEL; i += 256) gw[i] = gate_w[i];
  __syncthreads();
  int t = blockIdx.x * 256 + threadIdx.x;
  if (t >= T) return;
  float acc[NEXP];
#pragma unroll
  for (int e = 0; e < NEXP; ++e) acc[e] = 0.f;
  const float* xr = x + (size_t)t * DMODEL;
  for (int d = 0; d < DMODEL; d += 4) {
    v4f xv = *(const v4f*)(xr + d);
#pragma unroll
    for (int e = 0; e < NEXP; ++e) {
      const float* g = gw + e * DMODEL + d;
      acc[e] += xv.x * g[0] + xv.y * g[1] + xv.z * g[2] + xv.w * g[3];
    }
  }
  int e1 = 0; float l1 = acc[0];
#pragma unroll
  for (int e = 1; e < NEXP; ++e) if (acc[e] > l1) { l1 = acc[e]; e1 = e; }
  int e2 = -1; float l2 = -3.0e38f;
#pragma unroll
  for (int e = 0; e < NEXP; ++e) if (e != e1 && acc[e] > l2) { l2 = acc[e]; e2 = e; }
  float p = __expf(l2 - l1);          // softmax over the two selected logits
  float inv = 1.f / (1.f + p);
  tkIdx[2 * t] = e1; tkIdx[2 * t + 1] = e2;
  tkW[2 * t] = inv;  tkW[2 * t + 1] = p * inv;
  atomicAdd(&counts[e1], 1);
  atomicAdd(&counts[e2], 1);
}

// ---------------- tiny exclusive scan over 8 expert counts ----------------
__global__ void moe_scan(const int* __restrict__ counts, int* __restrict__ offsets) {
  if (threadIdx.x == 0) {
    int o = 0;
    for (int e = 0; e < NEXP; ++e) { offsets[e] = o; o += counts[e]; }
    offsets[NEXP] = o;
  }
}

// ---------------- scatter tokens into per-expert segments ----------------
__global__ __launch_bounds__(256) void moe_scatter(const int* __restrict__ tkIdx,
                                                   int* __restrict__ cursors,
                                                   const int* __restrict__ offsets,
                                                   int* __restrict__ rows,
                                                   int* __restrict__ invMap, int T) {
  int t = blockIdx.x * 256 + threadIdx.x;
  if (t >= T) return;
#pragma unroll
  for (int k = 0; k < 2; ++k) {
    int e = tkIdx[2 * t + k];
    int pos = atomicAdd(&cursors[e], 1);
    int slot = offsets[e] + pos;
    rows[slot] = t;
    invMap[2 * t + k] = slot;
  }
}

// ---------------- fp32 -> bf16 streaming convert (activations & weights) ------
__global__ __launch_bounds__(256) void moe_cvt_bf16(const float* __restrict__ src,
                                                    __bf16* __restrict__ dst, long n) {
  long i = ((long)blockIdx.x * 256 + threadIdx.x) * 8;
  if (i >= n) return;
  v8f v = *(const v8f*)(src + i);
  v8bf o;
#pragma unroll
  for (int j = 0; j < 8; ++j) o[j] = (__bf16)v[j];
  *(v8bf*)(dst + i) = o;
}

// ---------------- fragment helpers (contiguous k order, same for A and B) ----------------
// A-matrix 16x32 bf16: lane half h holds K = 16h..16h+15 (contiguous)
__device__ inline v16bf load_a_frag(const __bf16* row, int koff_half) {
  v8bf lo = *(const v8bf*)(row + koff_half);
  v8bf hi = *(const v8bf*)(row + koff_half + 8);
  v16bf o;
#pragma unroll
  for (int i = 0; i < 8; ++i) { o[i] = lo[i]; o[8 + i] = hi[i]; }
  return o;
}
// B-matrix 32x16 bf16: lane half h holds K = 16h..16h+15 (contiguous 32B)
__device__ inline v16bf load_b_frag(const __bf16* p) { return *(const v16bf*)p; }

// ---------------- grouped GEMM 1: h = relu(Xe @ W1[e]^T + b1[e]) ----------------
__global__ __launch_bounds__(256) void moe_ffn1(const __bf16* __restrict__ xbf,
                                                const __bf16* __restrict__ W1,
                                                const float* __restrict__ b1,
                                                const int* __restrict__ rows,
                                                const int* __restrict__ counts,
                                                const int* __restrict__ offsets,
                                                __bf16* __restrict__ h) {
  const int e = blockIdx.z;
  const int cnt = counts[e];
  const int mtile = blockIdx.x;
  if (mtile * 128 >= cnt) return;              // uniform exit: EXEC stays all-ones
  const int seg = offsets[e];
  const int lane = threadIdx.x & 31, wave = threadIdx.x >> 5;
  const int wm = wave & 1, wn = wave >> 1;     // 2x4 wave grid over 128x128 tile
  const int l15 = lane & 15, kh = lane >> 4;
  const int nBase = blockIdx.y * 128 + wn * 32;
  const size_t eW = (size_t)e * HID * DMODEL;

  const __bf16* aptr[4];
#pragma unroll
  for (int i = 0; i < 4; ++i) {
    int rl = mtile * 128 + wm * 64 + i * 16 + l15;
    int rc = rl < cnt ? rl : (cnt - 1);
    aptr[i] = xbf + (size_t)rows[seg + rc] * DMODEL;
  }
  const __bf16* bptr[2];
#pragma unroll
  for (int j = 0; j < 2; ++j)
    bptr[j] = W1 + eW + (size_t)(nBase + j * 16 + l15) * DMODEL;

  v8f acc[4][2];
#pragma unroll
  for (int i = 0; i < 4; ++i)
#pragma unroll
    for (int j = 0; j < 2; ++j) acc[i][j] = (v8f)(0.f);

  for (int k0 = 0; k0 < DMODEL; k0 += 32) {
    v16bf bf[2];
#pragma unroll
    for (int j = 0; j < 2; ++j) {
      bf[j] = load_b_frag(bptr[j] + k0 + 16 * kh);
      if (k0 + 256 < DMODEL) __builtin_prefetch(bptr[j] + k0 + 256, 0, 3);
    }
#pragma unroll
    for (int i = 0; i < 4; ++i) {
      v16bf a = load_a_frag(aptr[i], k0 + 16 * kh);
#pragma unroll
      for (int j = 0; j < 2; ++j)
        acc[i][j] = __builtin_amdgcn_wmma_f32_16x16x32_bf16(
            false, a, false, bf[j], (short)0, acc[i][j], false, false);
    }
  }
#pragma unroll
  for (int j = 0; j < 2; ++j) {
    int col = nBase + j * 16 + l15;
    float bv = b1[(size_t)e * HID + col];
#pragma unroll
    for (int i = 0; i < 4; ++i)
#pragma unroll
      for (int r = 0; r < 8; ++r) {
        int rl = mtile * 128 + wm * 64 + i * 16 + 8 * kh + r;
        if (rl < cnt) {
          float v = acc[i][j][r] + bv;
          h[(size_t)(seg + rl) * HID + col] = (__bf16)(v > 0.f ? v : 0.f);
        }
      }
  }
}

// ---------------- grouped GEMM 2: y = h @ W2[e]^T + b2[e] ----------------
__global__ __launch_bounds__(256) void moe_ffn2(const __bf16* __restrict__ h,
                                                const __bf16* __restrict__ W2,
                                                const float* __restrict__ b2,
                                                const int* __restrict__ counts,
                                                const int* __restrict__ offsets,
                                                float* __restrict__ y) {
  const int e = blockIdx.z;
  const int cnt = counts[e];
  const int mtile = blockIdx.x;
  if (mtile * 128 >= cnt) return;
  const int seg = offsets[e];
  const int lane = threadIdx.x & 31, wave = threadIdx.x >> 5;
  const int wm = wave & 1, wn = wave >> 1;
  const int l15 = lane & 15, kh = lane >> 4;
  const int nBase = blockIdx.y * 128 + wn * 32;
  const size_t eW = (size_t)e * DMODEL * HID;

  const __bf16* aptr[4];
#pragma unroll
  for (int i = 0; i < 4; ++i) {
    int rl = mtile * 128 + wm * 64 + i * 16 + l15;
    int rc = rl < cnt ? rl : (cnt - 1);
    aptr[i] = h + (size_t)(seg + rc) * HID;
  }
  const __bf16* bptr[2];
#pragma unroll
  for (int j = 0; j < 2; ++j)
    bptr[j] = W2 + eW + (size_t)(nBase + j * 16 + l15) * HID;

  v8f acc[4][2];
#pragma unroll
  for (int i = 0; i < 4; ++i)
#pragma unroll
    for (int j = 0; j < 2; ++j) acc[i][j] = (v8f)(0.f);

  for (int k0 = 0; k0 < HID; k0 += 32) {
    v16bf bf[2];
#pragma unroll
    for (int j = 0; j < 2; ++j) {
      bf[j] = load_b_frag(bptr[j] + k0 + 16 * kh);
      if (k0 + 256 < HID) __builtin_prefetch(bptr[j] + k0 + 256, 0, 3);
    }
#pragma unroll
    for (int i = 0; i < 4; ++i) {
      v16bf a = load_a_frag(aptr[i], k0 + 16 * kh);
#pragma unroll
      for (int j = 0; j < 2; ++j)
        acc[i][j] = __builtin_amdgcn_wmma_f32_16x16x32_bf16(
            false, a, false, bf[j], (short)0, acc[i][j], false, false);
    }
  }
#pragma unroll
  for (int j = 0; j < 2; ++j) {
    int col = nBase + j * 16 + l15;
    float bv = b2[(size_t)e * DMODEL + col];
#pragma unroll
    for (int i = 0; i < 4; ++i)
#pragma unroll
      for (int r = 0; r < 8; ++r) {
        int rl = mtile * 128 + wm * 64 + i * 16 + 8 * kh + r;
        if (rl < cnt) y[(size_t)(seg + rl) * DMODEL + col] = acc[i][j][r] + bv;
      }
  }
}

// ---------------- deterministic weighted combine ----------------
__global__ __launch_bounds__(256) void moe_combine(const float* __restrict__ y,
                                                   const int* __restrict__ invMap,
                                                   const float* __restrict__ tkW,
                                                   float* __restrict__ out, int T) {
  int t = blockIdx.x;
  if (t >= T) return;
  const float* y0 = y + (size_t)invMap[2 * t] * DMODEL;
  const float* y1 = y + (size_t)invMap[2 * t + 1] * DMODEL;
  float w0 = tkW[2 * t], w1 = tkW[2 * t + 1];
  int d = threadIdx.x * 4;
  v4f a = *(const v4f*)(y0 + d);
  v4f b = *(const v4f*)(y1 + d);
  v4f o = w0 * a + w1 * b;
  *(v4f*)(out + (size_t)t * DMODEL + d) = o;
}

extern "C" void kernel_launch(void* const* d_in, const int* in_sizes, int n_in,
                              void* d_out, int out_size, void* d_ws, size_t ws_size,
                              hipStream_t stream) {
  const float* x      = (const float*)d_in[0];
  const float* gate_w = (const float*)d_in[1];
  const float* W1     = (const float*)d_in[2];
  const float* b1     = (const float*)d_in[3];
  const float* W2     = (const float*)d_in[4];
  const float* b2     = (const float*)d_in[5];
  float* out = (float*)d_out;
  const int T = in_sizes[0] / DMODEL;           // 8192
  const long nW1 = (long)NEXP * HID * DMODEL;   // 33.5M
  const long nW2 = (long)NEXP * DMODEL * HID;   // 33.5M

  char* ws = (char*)d_ws;
  int*   counts  = (int*)(ws + 0);      // 8
  int*   cursors = (int*)(ws + 64);     // 8
  int*   offsets = (int*)(ws + 128);    // 9
  size_t off = 1024;
  int*   tkIdx = (int*)(ws + off);   off += (size_t)2 * T * 4;
  float* tkW   = (float*)(ws + off); off += (size_t)2 * T * 4;
  int*   rowsA = (int*)(ws + off);   off += (size_t)2 * T * 4;
  int*   invM  = (int*)(ws + off);   off += (size_t)2 * T * 4;
  off = (off + 255) & ~(size_t)255;
  __bf16* xbf  = (__bf16*)(ws + off); off += (size_t)T * DMODEL * 2;
  __bf16* w1bf = (__bf16*)(ws + off); off += (size_t)nW1 * 2;
  __bf16* w2bf = (__bf16*)(ws + off); off += (size_t)nW2 * 2;
  __bf16* hbf  = (__bf16*)(ws + off); off += (size_t)2 * T * HID * 2;
  off = (off + 255) & ~(size_t)255;
  float* ybuf  = (float*)(ws + off);  off += (size_t)2 * T * DMODEL * 4;

  hipMemsetAsync(ws, 0, 1024, stream);  // counts/cursors/offsets
  moe_router<<<(T + 255) / 256, 256, 0, stream>>>(x, gate_w, counts, tkIdx, tkW, T);
  moe_scan<<<1, 32, 0, stream>>>(counts, offsets);
  moe_scatter<<<(T + 255) / 256, 256, 0, stream>>>(tkIdx, cursors, offsets, rowsA, invM, T);

  long nX = (long)T * DMODEL;
  moe_cvt_bf16<<<(unsigned)((nX / 8 + 255) / 256), 256, 0, stream>>>(x, xbf, nX);
  moe_cvt_bf16<<<(unsigned)((nW1 / 8 + 255) / 256), 256, 0, stream>>>(W1, w1bf, nW1);
  moe_cvt_bf16<<<(unsigned)((nW2 / 8 + 255) / 256), 256, 0, stream>>>(W2, w2bf, nW2);

  dim3 g1((T + 127) / 128, HID / 128, NEXP);
  moe_ffn1<<<g1, 256, 0, stream>>>(xbf, w1bf, b1, rowsA, counts, offsets, hbf);
  dim3 g2((T + 127) / 128, DMODEL / 128, NEXP);
  moe_ffn2<<<g2, 256, 0, stream>>>(hbf, w2bf, b2, counts, offsets, ybuf);
  moe_combine<<<T, 256, 0, stream>>>(ybuf, invM, tkW, out, T);
}
